// MUSIQ_Encoder_63505386439238
// MI455X (gfx1250) — matrix operand order repaired and checked by reading, not compiled, not run on hardware
//
#include <hip/hip_runtime.h>
#include <hip/hip_bf16.h>

typedef __attribute__((ext_vector_type(16))) _Float16 v16h;
typedef __attribute__((ext_vector_type(8)))  _Float16 h8;
typedef __attribute__((ext_vector_type(8)))  float    v8f;
typedef __attribute__((ext_vector_type(4)))  _Float16 hvec4;
typedef int v4i32 __attribute__((vector_size(16)));

#define B_   32
#define H_   384
#define NH_  6
#define DH_  64
#define DF_  1536
#define L_   14
#define S_   770
#define SP_  784              // 49 * 16, padded sequence
#define MTOT (B_ * SP_)       // 25088 rows, multiple of 64

#if defined(__gfx1250__) && \
    __has_builtin(__builtin_amdgcn_global_load_async_to_lds_b128) && \
    __has_builtin(__builtin_amdgcn_s_wait_asynccnt)
#define USE_ASYNC_COPY 1
#else
#define USE_ASYNC_COPY 0
#endif

static __device__ __forceinline__ v8f wmma16(v16h a, v16h b, v8f c) {
  return __builtin_amdgcn_wmma_f32_16x16x32_f16(false, a, false, b, (short)0, c,
                                                false, false);
}

// Build a 16-half fragment from two contiguous 8-half (16B) spans -> 2x b128.
static __device__ __forceinline__ v16h ldfrag(const _Float16* p0,
                                              const _Float16* p1) {
  h8 lo = *reinterpret_cast<const h8*>(p0);
  h8 hi = *reinterpret_cast<const h8*>(p1);
  return __builtin_shufflevector(lo, hi, 0, 1, 2, 3, 4, 5, 6, 7, 8, 9, 10, 11,
                                 12, 13, 14, 15);
}

// Stage 16 bytes global -> LDS. Async (ASYNCcnt) when available.
static __device__ __forceinline__ void stage_b128(const _Float16* g,
                                                  _Float16* l) {
#if USE_ASYNC_COPY
  __builtin_amdgcn_global_load_async_to_lds_b128((v4i32*)g, (v4i32*)l, 0, 0);
#else
  *reinterpret_cast<h8*>(l) = *reinterpret_cast<const h8*>(g);
#endif
}

static __device__ __forceinline__ void stage_wait() {
#if USE_ASYNC_COPY
  __builtin_amdgcn_s_wait_asynccnt(0);
#endif
}

// ------------------------------------------------------- weight swizzle
// Per (kt,nt) tile, per lane, 16 contiguous halves in this file's contiguous-k B-fragment
// order: b[e] = W[kt*32 + (lane>>4)*16 + e][nt*16 + (lane&15)].
__global__ void swizzle_weights(const float* __restrict__ W,
                                _Float16* __restrict__ out, int K, int N) {
  int idx = blockIdx.x * 256 + threadIdx.x;   // exact grid over L*K*N
  int per = K * N;
  int l = idx / per;
  int r = idx % per;
  int e = r & 15;
  int lane = (r >> 4) & 31;
  int rest = r >> 9;
  int ntiles = N >> 4;
  int nt = rest % ntiles;
  int kt = rest / ntiles;
  int kk = kt * 32 + (lane >> 4) * 16 + e;
  int n = nt * 16 + (lane & 15);
  out[idx] = (_Float16)W[(size_t)l * per + (size_t)kk * N + n];
}

// ---------------------------------------------------------------- embeddings
__global__ void embed_kernel(const float* __restrict__ org,
                             const float* __restrict__ s1e,
                             const float* __restrict__ s2e,
                             const float* __restrict__ sce0,
                             const float* __restrict__ sce1,
                             const float* __restrict__ sce2,
                             const float* __restrict__ hse,
                             const float* __restrict__ cls,
                             float* __restrict__ xf,
                             _Float16* __restrict__ xh) {
  int idx = blockIdx.x * 256 + threadIdx.x;   // exact grid: MTOT*H_/256
  int h = idx % H_;
  int t = idx / H_;
  int s = t % SP_;
  int b = t / SP_;
  float val = 0.f;
  if (s == 0) {
    val = cls[h];
  } else if (s < 577) {
    int p = s - 1, i = p / 24, j = p % 24;
    int ti = (i * 10) / 24, tj = (j * 10) / 24;
    val = org[((b * H_ + h) * 24 + i) * 24 + j] + sce0[h] +
          hse[(h * 10 + ti) * 10 + tj];
  } else if (s < 721) {
    int p = s - 577, i = p / 12, j = p % 12;
    int ti = (i * 10) / 12, tj = (j * 10) / 12;
    val = s1e[((b * H_ + h) * 12 + i) * 12 + j] + sce1[h] +
          hse[(h * 10 + ti) * 10 + tj];
  } else if (s < 770) {
    int p = s - 721, i = p / 7, j = p % 7;
    int ti = (i * 10) / 7, tj = (j * 10) / 7;
    val = s2e[((b * H_ + h) * 7 + i) * 7 + j] + sce2[h] +
          hse[(h * 10 + ti) * 10 + tj];
  }
  size_t o = (size_t)(b * SP_ + s) * H_ + h;
  xf[o] = val;
  xh[o] = (_Float16)val;
}

// ---------------------------------------------------------------- WMMA GEMM
// C[M,N] = act(A[M,K] * W[K,N] + bias)(+resid). A async-staged in LDS, W read
// directly from pre-swizzled global (L2-resident) in fragment order.
__global__ __launch_bounds__(128) void gemm_f16_kernel(
    const _Float16* __restrict__ A, const _Float16* __restrict__ Wswz,
    const float* __restrict__ bias, const float* __restrict__ resid,
    float* __restrict__ outF, _Float16* __restrict__ outH,
    int K, int N, int gelu) {
  __shared__ alignas(16) _Float16 As[64][40];   // 64 x 32 (+pad)
  int tid = threadIdx.x;
  int wave = tid >> 5, lane = tid & 31;
  int half = lane >> 4, nn = lane & 15;
  int row0 = blockIdx.x * 64, col0 = blockIdx.y * 64;
  int ntiles = N >> 4;
  int nt0 = blockIdx.y * 4;

  v8f acc[4];
#pragma unroll
  for (int t = 0; t < 4; t++)
#pragma unroll
    for (int i = 0; i < 8; i++) acc[t][i] = 0.f;

  for (int kb = 0, kt = 0; kb < K; kb += 32, kt++) {
#pragma unroll
    for (int i = 0; i < 2; i++) {               // 256 16B groups: 64 rows x 4
      int g = tid + i * 128;
      int r = g >> 2, c8 = (g & 3) * 8;
      stage_b128(A + (size_t)(row0 + r) * K + kb + c8, &As[r][c8]);
    }
    if (kb + 32 < K)                            // global_prefetch_b8
      __builtin_prefetch(A + (size_t)(row0 + (tid & 63)) * K + kb + 32, 0, 1);
    stage_wait();
    __syncthreads();

    v16h a = ldfrag(&As[wave * 16 + nn][half * 16],
                    &As[wave * 16 + nn][half * 16 + 8]);
    const _Float16* wbase =
        Wswz + (((size_t)kt * ntiles + nt0) * 32 + lane) * 16;
#pragma unroll
    for (int t = 0; t < 4; t++) {
      v16h b = *reinterpret_cast<const v16h*>(wbase + (size_t)t * 512);
      acc[t] = wmma16(a, b, acc[t]);
    }
    __syncthreads();
  }

  // C/D layout: VGPR r -> row r + 8*half, col = lane&15.
#pragma unroll
  for (int t = 0; t < 4; t++) {
#pragma unroll
    for (int r = 0; r < 8; r++) {
      int grow = row0 + wave * 16 + half * 8 + r;
      int gcol = col0 + t * 16 + nn;
      float val = acc[t][r];
      if (bias)  val += bias[gcol];
      if (gelu)  val = 0.5f * val * (1.f + erff(val * 0.70710678118f));
      if (resid) val += resid[(size_t)grow * N + gcol];
      if (outF)  outF[(size_t)grow * N + gcol] = val;
      if (outH)  outH[(size_t)grow * N + gcol] = (_Float16)val;
    }
  }
}

// ---------------------------------------------------------- flash attention
// Block: 128 thr = 4 waves, each wave owns one 16-row q-tile; the 4 waves
// share K/V blocks staged in LDS. Online softmax over 32-key blocks.
__global__ __launch_bounds__(128) void attn_kernel(
    const _Float16* __restrict__ q, const _Float16* __restrict__ k,
    const _Float16* __restrict__ v, _Float16* __restrict__ ctx) {
  __shared__ alignas(16) _Float16 Qs[4][16][72];
  __shared__ alignas(16) _Float16 Ks[32][72];    // [key][dim]
  __shared__ alignas(16) _Float16 Vst[64][40];   // [dim][key] (transposed)
  __shared__ alignas(16) _Float16 Ps[4][16][40];
  int tid = threadIdx.x;
  int wave = tid >> 5, lane = tid & 31;
  int half = lane >> 4, nn = lane & 15;
  int qt = blockIdx.x * 4 + wave;
  if (qt > 48) qt = 48;                          // wave-uniform clamp
  int hd = blockIdx.y, b = blockIdx.z;
  size_t base = (size_t)b * SP_ * 384 + hd * 64;

  // Per-wave Q tile (16x64) async-staged; scale applied to scores later.
#pragma unroll
  for (int i = 0; i < 4; i++) {
    int g = lane + i * 32;            // 128 16B groups: 16 rows x 8
    int r = g >> 3, c8 = (g & 7) * 8;
    stage_b128(q + base + (size_t)(qt * 16 + r) * 384 + c8, &Qs[wave][r][c8]);
  }
  stage_wait();
  __syncthreads();

  v16h aq0 = ldfrag(&Qs[wave][nn][half * 16], &Qs[wave][nn][half * 16 + 8]);
  v16h aq1 = ldfrag(&Qs[wave][nn][32 + half * 16], &Qs[wave][nn][32 + half * 16 + 8]);

  float mrow[8], ssum[8];
  v8f o[4];
#pragma unroll
  for (int r = 0; r < 8; r++) { mrow[r] = -1e30f; ssum[r] = 0.f; }
#pragma unroll
  for (int t = 0; t < 4; t++)
#pragma unroll
    for (int i = 0; i < 8; i++) o[t][i] = 0.f;

  for (int kbi = 0; kbi < 25; kbi++) {   // 25*32 = 800 >= 770 keys
    int key0 = kbi * 32;
    __syncthreads();                     // LDS reuse guard
#pragma unroll
    for (int i = 0; i < 2; i++) {        // K rows: 256 16B groups
      int g = tid + i * 128;
      int r = g >> 3, c8 = (g & 7) * 8;
      int kg = key0 + r;
      if (kg < SP_) {
        stage_b128(k + base + (size_t)kg * 384 + c8, &Ks[r][c8]);
      } else {
        h8 z;
#pragma unroll
        for (int j = 0; j < 8; j++) z[j] = (_Float16)0.f;
        *reinterpret_cast<h8*>(&Ks[r][c8]) = z;
      }
    }
#pragma unroll
    for (int i = 0; i < 4; i++) {        // V rows -> transposed scatter
      int g = tid + i * 128;
      int r = g >> 4, c4 = (g & 15) * 4;
      int kg = key0 + r;
      hvec4 vv4;
      if (kg < SP_) {
        vv4 = *reinterpret_cast<const hvec4*>(v + base + (size_t)kg * 384 + c4);
      } else {
#pragma unroll
        for (int j = 0; j < 4; j++) vv4[j] = (_Float16)0.f;
      }
#pragma unroll
      for (int j = 0; j < 4; j++) Vst[c4 + j][r] = vv4[j];
    }
    stage_wait();
    __syncthreads();

    // scores: S = (Q * K^T) * 0.125 (two 16-key subtiles, 2 WMMAs each)
    v8f sc[2];
#pragma unroll
    for (int sub = 0; sub < 2; sub++) {
      v8f z;
#pragma unroll
      for (int i = 0; i < 8; i++) z[i] = 0.f;
      v16h bk0 = ldfrag(&Ks[sub * 16 + nn][half * 16],
                        &Ks[sub * 16 + nn][half * 16 + 8]);
      z = wmma16(aq0, bk0, z);
      v16h bk1 = ldfrag(&Ks[sub * 16 + nn][32 + half * 16],
                        &Ks[sub * 16 + nn][32 + half * 16 + 8]);
      z = wmma16(aq1, bk1, z);
      sc[sub] = z;
    }

    int keyA = key0 + nn, keyB = key0 + 16 + nn;
#pragma unroll
    for (int r = 0; r < 8; r++) {
      float s0 = (keyA < S_) ? sc[0][r] * 0.125f : -1e30f;
      float s1 = (keyB < S_) ? sc[1][r] * 0.125f : -1e30f;
      float rm = fmaxf(s0, s1);
#pragma unroll
      for (int off = 1; off < 16; off <<= 1)
        rm = fmaxf(rm, __shfl_xor(rm, off, 32));
      float mnew = fmaxf(mrow[r], rm);
      float p0 = __expf(s0 - mnew);
      float p1 = __expf(s1 - mnew);
      float rs = p0 + p1;
#pragma unroll
      for (int off = 1; off < 16; off <<= 1) rs += __shfl_xor(rs, off, 32);
      float alpha = __expf(mrow[r] - mnew);
      ssum[r] = ssum[r] * alpha + rs;
      mrow[r] = mnew;
#pragma unroll
      for (int t = 0; t < 4; t++) o[t][r] *= alpha;
      int prow = half * 8 + r;
      Ps[wave][prow][nn]      = (_Float16)p0;
      Ps[wave][prow][16 + nn] = (_Float16)p1;
    }
    __syncthreads();

    // O += P(16x32) * V(32x64)
    v16h ap = ldfrag(&Ps[wave][nn][half * 16], &Ps[wave][nn][half * 16 + 8]);
#pragma unroll
    for (int t = 0; t < 4; t++) {
      v16h bv = ldfrag(&Vst[t * 16 + nn][half * 16],
                       &Vst[t * 16 + nn][half * 16 + 8]);
      o[t] = wmma16(ap, bv, o[t]);
    }
  }

#pragma unroll
  for (int t = 0; t < 4; t++) {
#pragma unroll
    for (int r = 0; r < 8; r++) {
      int row = qt * 16 + half * 8 + r;
      float val = o[t][r] / ssum[r];
      ctx[base + (size_t)row * 384 + t * 16 + nn] = (_Float16)val;
    }
  }
}

// ---------------------------------------------------------------- layernorm
__global__ __launch_bounds__(128) void layernorm_kernel(
    const float* __restrict__ in, const float* __restrict__ g,
    const float* __restrict__ be, float* __restrict__ outF,
    _Float16* __restrict__ outH) {
  __shared__ float s1[128], s2[128];
  int row = blockIdx.x, tid = threadIdx.x;
  const float* r = in + (size_t)row * H_;
  float x0 = r[tid], x1 = r[tid + 128], x2 = r[tid + 256];
  s1[tid] = x0 + x1 + x2;
  s2[tid] = x0 * x0 + x1 * x1 + x2 * x2;
  __syncthreads();
  for (int off = 64; off > 0; off >>= 1) {
    if (tid < off) { s1[tid] += s1[tid + off]; s2[tid] += s2[tid + off]; }
    __syncthreads();
  }
  float mean = s1[0] * (1.f / H_);
  float var = s2[0] * (1.f / H_) - mean * mean;
  float rstd = rsqrtf(var + 1e-6f);
#pragma unroll
  for (int i = 0; i < 3; i++) {
    int e = tid + i * 128;
    float y = (r[e] - mean) * rstd * g[e] + be[e];
    outF[(size_t)row * H_ + e] = y;
    outH[(size_t)row * H_ + e] = (_Float16)y;
  }
}

// ---------------------------------------------------------------- final copy
__global__ void final_copy(const float* __restrict__ xf,
                           float* __restrict__ out) {
  int idx = blockIdx.x * 256 + threadIdx.x;   // exact grid over B*S*H
  int h = idx % H_;
  int t = idx / H_;
  int s = t % S_;
  int b = t / S_;
  out[idx] = xf[(size_t)(b * SP_ + s) * H_ + h];
}

// ---------------------------------------------------------------- launcher
extern "C" void kernel_launch(void* const* d_in, const int* in_sizes, int n_in,
                              void* d_out, int out_size, void* d_ws,
                              size_t ws_size, hipStream_t stream) {
  (void)in_sizes; (void)n_in; (void)out_size; (void)ws_size;
  const float* org  = (const float*)d_in[0];
  const float* s1e  = (const float*)d_in[1];
  const float* s2e  = (const float*)d_in[2];
  const float* sce0 = (const float*)d_in[3];
  const float* sce1 = (const float*)d_in[4];
  const float* sce2 = (const float*)d_in[5];
  const float* hse  = (const float*)d_in[6];
  const float* cls  = (const float*)d_in[7];
  const float* Wq = (const float*)d_in[8];  const float* bq = (const float*)d_in[9];
  const float* Wk = (const float*)d_in[10]; const float* bk = (const float*)d_in[11];
  const float* Wv = (const float*)d_in[12]; const float* bv = (const float*)d_in[13];
  const float* Wo = (const float*)d_in[14]; const float* bo = (const float*)d_in[15];
  const float* g1 = (const float*)d_in[16]; const float* be1 = (const float*)d_in[17];
  const float* W1 = (const float*)d_in[18]; const float* b1 = (const float*)d_in[19];
  const float* W2 = (const float*)d_in[20]; const float* b2 = (const float*)d_in[21];
  const float* g2 = (const float*)d_in[22]; const float* be2 = (const float*)d_in[23];

  char* ws = (char*)d_ws;
  size_t off = 0;
  auto take = [&](size_t bytes) -> void* {
    void* p = (void*)(ws + off);
    off = (off + bytes + 255) & ~(size_t)255;
    return p;
  };
  const size_t nWp = (size_t)L_ * H_ * 384;
  const size_t nWf = (size_t)L_ * H_ * DF_;
  _Float16* Wqh = (_Float16*)take(nWp * 2);
  _Float16* Wkh = (_Float16*)take(nWp * 2);
  _Float16* Wvh = (_Float16*)take(nWp * 2);
  _Float16* Woh = (_Float16*)take(nWp * 2);
  _Float16* W1h = (_Float16*)take(nWf * 2);
  _Float16* W2h = (_Float16*)take(nWf * 2);
  float*    xf  = (float*)take((size_t)MTOT * H_ * 4);
  float*    xr  = (float*)take((size_t)MTOT * H_ * 4);
  _Float16* xh  = (_Float16*)take((size_t)MTOT * H_ * 2);
  _Float16* qb  = (_Float16*)take((size_t)MTOT * 384 * 2);
  _Float16* kbf = (_Float16*)take((size_t)MTOT * 384 * 2);
  _Float16* vbf = (_Float16*)take((size_t)MTOT * 384 * 2);
  _Float16* cbf = (_Float16*)take((size_t)MTOT * 384 * 2);
  _Float16* hbf = (_Float16*)take((size_t)MTOT * DF_ * 2);

  // swizzle all weights into WMMA B-fragment order (one-time)
  swizzle_weights<<<(int)(nWp / 256), 256, 0, stream>>>(Wq, Wqh, H_, 384);
  swizzle_weights<<<(int)(nWp / 256), 256, 0, stream>>>(Wk, Wkh, H_, 384);
  swizzle_weights<<<(int)(nWp / 256), 256, 0, stream>>>(Wv, Wvh, H_, 384);
  swizzle_weights<<<(int)(nWp / 256), 256, 0, stream>>>(Wo, Woh, 384, H_);
  swizzle_weights<<<(int)(nWf / 256), 256, 0, stream>>>(W1, W1h, H_, DF_);
  swizzle_weights<<<(int)(nWf / 256), 256, 0, stream>>>(W2, W2h, DF_, H_);

  embed_kernel<<<(MTOT * H_) / 256, 256, 0, stream>>>(
      org, s1e, s2e, sce0, sce1, sce2, hse, cls, xf, xh);

  dim3 gP(MTOT / 64, 384 / 64);   // (392, 6)
  dim3 gF(MTOT / 64, DF_ / 64);   // (392, 24)
  dim3 gA(13, NH_, B_);           // 13*4 waves cover 49 q-tiles

  for (int l = 0; l < L_; l++) {
    const _Float16* wq = Wqh + (size_t)l * H_ * 384;
    const _Float16* wk = Wkh + (size_t)l * H_ * 384;
    const _Float16* wv = Wvh + (size_t)l * H_ * 384;
    const _Float16* wo = Woh + (size_t)l * H_ * 384;
    const _Float16* w1 = W1h + (size_t)l * H_ * DF_;
    const _Float16* w2 = W2h + (size_t)l * DF_ * H_;

    gemm_f16_kernel<<<gP, 128, 0, stream>>>(xh, wq, bq + l * 384, nullptr,
                                            nullptr, qb, H_, 384, 0);
    gemm_f16_kernel<<<gP, 128, 0, stream>>>(xh, wk, bk + l * 384, nullptr,
                                            nullptr, kbf, H_, 384, 0);
    gemm_f16_kernel<<<gP, 128, 0, stream>>>(xh, wv, bv + l * 384, nullptr,
                                            nullptr, vbf, H_, 384, 0);
    attn_kernel<<<gA, 128, 0, stream>>>(qb, kbf, vbf, cbf);
    gemm_f16_kernel<<<gP, 128, 0, stream>>>(cbf, wo, bo + l * H_, xf, xr,
                                            nullptr, 384, H_, 0);
    layernorm_kernel<<<MTOT, 128, 0, stream>>>(xr, g1 + l * H_, be1 + l * H_,
                                               xf, xh);
    gemm_f16_kernel<<<gF, 128, 0, stream>>>(xh, w1, b1 + l * DF_, nullptr,
                                            nullptr, hbf, H_, DF_, 1);
    gemm_f16_kernel<<<gP, 128, 0, stream>>>(hbf, w2, b2 + l * H_, xf, xr,
                                            nullptr, DF_, H_, 0);
    layernorm_kernel<<<MTOT, 128, 0, stream>>>(xr, g2 + l * H_, be2 + l * H_,
                                               xf, xh);
  }

  final_copy<<<(B_ * S_ * H_) / 256, 256, 0, stream>>>(xf, (float*)d_out);
}
